// SerializedMamba_30305289240961
// MI455X (gfx1250) — matrix operand order repaired and checked by reading, compiled, not run on hardware
//
#include <hip/hip_runtime.h>

#define NPTS   16384
#define CDIM   512
#define DI     1024
#define DSTATE 16
#define DTR    32
#define NXP    64     // DT_RANK + 2*D_STATE
#define MLPD   1024
#define CHUNKS 128
#define CLEN   128    // CHUNKS*CLEN == NPTS

typedef float  v8f   __attribute__((ext_vector_type(8)));
typedef __bf16 v16bf __attribute__((ext_vector_type(16)));
typedef __bf16 v8bf  __attribute__((ext_vector_type(8)));
typedef int    v4i   __attribute__((ext_vector_type(4)));

// ---------------------------------------------------------------------------
// CDNA5 async global->LDS (ASYNCcnt) with safe fallback.
// Probe-confirmed signature: (AS1 int4* gsrc, AS3 int4* ldst, imm offset, imm cpol)
// ---------------------------------------------------------------------------
#if defined(__AMDGCN__) && __has_builtin(__builtin_amdgcn_global_load_async_to_lds_b128)
#define USE_ASYNC_LDS 1
#else
#define USE_ASYNC_LDS 0
#endif

#if USE_ASYNC_LDS
#define G2L_B128(gsrc, ldst)                                                \
  __builtin_amdgcn_global_load_async_to_lds_b128(                           \
      (__attribute__((address_space(1))) v4i*)(gsrc),                       \
      (__attribute__((address_space(3))) v4i*)(ldst), 0, 0)
#endif

__device__ __forceinline__ void wait_async0() {
#if defined(__AMDGCN__)
#if __has_builtin(__builtin_amdgcn_s_wait_asynccnt)
  __builtin_amdgcn_s_wait_asynccnt(0);
#else
  asm volatile("s_wait_asynccnt 0x0" ::: "memory");
#endif
#endif
}

// ---------------------------------------------------------------------------
// Weight conversion: f32 -> bf16 (optionally transposed to [N,K] layout)
// ---------------------------------------------------------------------------
__global__ __launch_bounds__(256) void cvt_bf_k(const float* __restrict__ s,
                                                __bf16* __restrict__ d, int n) {
  int i = blockIdx.x * 256 + threadIdx.x;
  if (i < n) d[i] = (__bf16)s[i];
}

// src [R][Cc] row-major  ->  dst [Cc][R]  (bf16)
__global__ __launch_bounds__(256) void cvt_bfT_k(const float* __restrict__ s,
                                                 __bf16* __restrict__ d, int R, int Cc) {
  int i = blockIdx.x * 256 + threadIdx.x;
  if (i < R * Cc) {
    int r = i / Cc, c = i % Cc;
    d[(size_t)c * R + r] = (__bf16)s[i];
  }
}

// ---------------------------------------------------------------------------
// LayerNorm over rows of width 512 (block=256, 2 elems/thread).
// GATHER=true: row = order[n]; also stores f32 copy (residual).
// ---------------------------------------------------------------------------
template <bool GATHER>
__global__ __launch_bounds__(256) void ln512_k(const float* __restrict__ src,
                                               const int* __restrict__ order,
                                               const float* __restrict__ w,
                                               const float* __restrict__ b,
                                               float* __restrict__ fcopy,
                                               __bf16* __restrict__ outbf) {
  const int n = blockIdx.x, tid = threadIdx.x;
  const int row = GATHER ? order[n] : n;
  const float* p = src + (size_t)row * CDIM;
  float v0 = p[tid], v1 = p[tid + 256];
  if (GATHER) {
    fcopy[(size_t)n * CDIM + tid] = v0;
    fcopy[(size_t)n * CDIM + tid + 256] = v1;
  }
  float s = v0 + v1, q = v0 * v0 + v1 * v1;
#pragma unroll
  for (int m = 16; m >= 1; m >>= 1) { s += __shfl_xor(s, m); q += __shfl_xor(q, m); }
  __shared__ float ss[8], qq[8];
  int wid = tid >> 5;
  if ((tid & 31) == 0) { ss[wid] = s; qq[wid] = q; }
  __syncthreads();
  float S = 0.f, Q = 0.f;
#pragma unroll
  for (int i = 0; i < 8; ++i) { S += ss[i]; Q += qq[i]; }
  const float mu = S * (1.f / CDIM);
  const float var = Q * (1.f / CDIM) - mu * mu;
  const float r = rsqrtf(var + 1e-5f);
  outbf[(size_t)n * CDIM + tid]       = (__bf16)((v0 - mu) * r * w[tid] + b[tid]);
  outbf[(size_t)n * CDIM + tid + 256] = (__bf16)((v1 - mu) * r * w[tid + 256] + b[tid + 256]);
}

// ---------------------------------------------------------------------------
// bf16 WMMA GEMM:  out[M,Nout] = A[M,K](bf16) x W[Nout,K](bf16)^T, f32 accum.
// block = 256 threads = 8 wave32; wave tile 32x64 = 2x4 wmma 16x16x32 frags.
// Double-buffered LDS; K-tile k+1 fetched via async global->LDS (ASYNCcnt)
// while computing K-tile k.
// EPI: 0 store f32 | 1 res+acc f32 | 2 gelu(acc+bias)->bf16 | 3 res+acc+bias, scatter by idx
// ---------------------------------------------------------------------------
template <int BM, int BN, int EPI>
__global__ __launch_bounds__(256) void gemm_bf16_k(const __bf16* __restrict__ A,
                                                   const __bf16* __restrict__ W,
                                                   int K, int Nout,
                                                   float* __restrict__ outF,
                                                   __bf16* __restrict__ outB,
                                                   const float* __restrict__ res,
                                                   const float* __restrict__ bias,
                                                   const int* __restrict__ idx) {
  constexpr int LDA = 40;          // bf16 row stride: 80B keeps 16B align, spreads banks
  constexpr int WROWS = BM / 32;
  constexpr int WCOLS = BN / 64;
  constexpr int AV = (BM * 4) / 256;   // uint4 staging vectors per thread (A tile)
  constexpr int BV = (BN * 4) / 256;
  static_assert(WROWS * WCOLS == 8, "8 waves per block");
  __shared__ __bf16 Als[2][BM * LDA];
  __shared__ __bf16 Bls[2][BN * LDA];

  const int tid = threadIdx.x;
  const int lane = tid & 31;
  const int wid = tid >> 5;
  const int wrow = wid % WROWS;
  const int wcol = wid / WROWS;
  const int m0 = blockIdx.y * BM;
  const int n0 = blockIdx.x * BN;
  const int half = lane >> 4;
  const int l15 = lane & 15;

  // per-thread staging coordinates (tile-local)
  int arow[AV], acol[AV], brow[BV], bcol[BV];
#pragma unroll
  for (int j = 0; j < AV; ++j) { int v = tid + j * 256; arow[j] = v >> 2; acol[j] = (v & 3) << 3; }
#pragma unroll
  for (int j = 0; j < BV; ++j) { int v = tid + j * 256; brow[j] = v >> 2; bcol[j] = (v & 3) << 3; }

#if !USE_ASYNC_LDS
  uint4 ar[AV], br[BV];
#endif

  auto stage_issue = [&](int kt, int buf) {
    const int k0 = kt * 32;
#if USE_ASYNC_LDS
#pragma unroll
    for (int j = 0; j < AV; ++j)
      G2L_B128(&A[(size_t)(m0 + arow[j]) * K + k0 + acol[j]],
               &Als[buf][arow[j] * LDA + acol[j]]);
#pragma unroll
    for (int j = 0; j < BV; ++j)
      G2L_B128(&W[(size_t)(n0 + brow[j]) * K + k0 + bcol[j]],
               &Bls[buf][brow[j] * LDA + bcol[j]]);
#else
    (void)buf;
#pragma unroll
    for (int j = 0; j < AV; ++j)
      ar[j] = *reinterpret_cast<const uint4*>(&A[(size_t)(m0 + arow[j]) * K + k0 + acol[j]]);
#pragma unroll
    for (int j = 0; j < BV; ++j)
      br[j] = *reinterpret_cast<const uint4*>(&W[(size_t)(n0 + brow[j]) * K + k0 + bcol[j]]);
#endif
  };
  auto stage_commit = [&](int buf) {
#if USE_ASYNC_LDS
    (void)buf;
    wait_async0();        // this wave's async-to-LDS writes have landed
#else
#pragma unroll
    for (int j = 0; j < AV; ++j)
      *reinterpret_cast<uint4*>(&Als[buf][arow[j] * LDA + acol[j]]) = ar[j];
#pragma unroll
    for (int j = 0; j < BV; ++j)
      *reinterpret_cast<uint4*>(&Bls[buf][brow[j] * LDA + bcol[j]]) = br[j];
#endif
  };

  v8f acc[2][4] = {};
  const int NK = K >> 5;

  stage_issue(0, 0);
  stage_commit(0);
  __syncthreads();

  for (int kt = 0; kt < NK; ++kt) {
    const int cur = kt & 1;
    const int nxt = cur ^ 1;
    if (kt + 1 < NK) stage_issue(kt + 1, nxt);   // overlap HBM fetch with WMMA

    v16bf afrag[2], bfrag[4];
#pragma unroll
    for (int mt = 0; mt < 2; ++mt) {   // A 16x32: lane=M, K = {half*8..+7, 16+half*8..+7}
      int r = wrow * 32 + mt * 16 + l15;
      v8bf lo = *reinterpret_cast<const v8bf*>(&Als[cur][r * LDA + half * 8]);
      v8bf hi = *reinterpret_cast<const v8bf*>(&Als[cur][r * LDA + 16 + half * 8]);
      afrag[mt] = __builtin_shufflevector(lo, hi, 0, 1, 2, 3, 4, 5, 6, 7,
                                          8, 9, 10, 11, 12, 13, 14, 15);
    }
#pragma unroll
    for (int nt = 0; nt < 4; ++nt) {   // B 32x16: lane=N, K = {half*8..+7, 16+half*8..+7}
      int r = wcol * 64 + nt * 16 + l15;
      v8bf lo = *reinterpret_cast<const v8bf*>(&Bls[cur][r * LDA + half * 8]);
      v8bf hi = *reinterpret_cast<const v8bf*>(&Bls[cur][r * LDA + 16 + half * 8]);
      bfrag[nt] = __builtin_shufflevector(lo, hi, 0, 1, 2, 3, 4, 5, 6, 7,
                                          8, 9, 10, 11, 12, 13, 14, 15);
    }
#pragma unroll
    for (int mt = 0; mt < 2; ++mt)
#pragma unroll
      for (int nt = 0; nt < 4; ++nt)
        acc[mt][nt] = __builtin_amdgcn_wmma_f32_16x16x32_bf16(
            false, afrag[mt], false, bfrag[nt], (short)0, acc[mt][nt], false, false);

    if (kt + 1 < NK) {
      stage_commit(nxt);
      __syncthreads();
    }
  }

  // Epilogue. C/D layout: lane=N (l15), M = half*8 + vgpr_e (per 16x16 subtile).
#pragma unroll
  for (int mt = 0; mt < 2; ++mt) {
#pragma unroll
    for (int nt = 0; nt < 4; ++nt) {
      const int n = n0 + wcol * 64 + nt * 16 + l15;
      const int mb = m0 + wrow * 32 + mt * 16 + half * 8;
#pragma unroll
      for (int e = 0; e < 8; ++e) {
        const int m = mb + e;
        float v = acc[mt][nt][e];
        if constexpr (EPI == 0) {
          outF[(size_t)m * Nout + n] = v;
        } else if constexpr (EPI == 1) {
          outF[(size_t)m * Nout + n] = res[(size_t)m * Nout + n] + v;
        } else if constexpr (EPI == 2) {
          float t = v + bias[n];
          float g = 0.5f * t * (1.f + erff(t * 0.70710678118654752f));
          outB[(size_t)m * Nout + n] = (__bf16)g;
        } else {
          float t = res[(size_t)m * Nout + n] + v + bias[n];
          outF[(size_t)idx[m] * Nout + n] = t;   // fused inverse-permutation scatter
        }
      }
    }
  }
}

// ---------------------------------------------------------------------------
// Depthwise causal conv (k=4) + SiLU. Reads xm = xz[:, :DI]. Writes f32 + bf16.
// ---------------------------------------------------------------------------
__global__ __launch_bounds__(256) void conv_silu_k(const float* __restrict__ xz,
                                                   const float* __restrict__ cw,
                                                   const float* __restrict__ cb,
                                                   float* __restrict__ xc,
                                                   __bf16* __restrict__ xc_bf) {
  const int i = blockIdx.x * 256 + threadIdx.x;
  const int d = i & (DI - 1);
  const int n = i >> 10;
  float a = cb[d];
#pragma unroll
  for (int k = 0; k < 4; ++k) {
    int t = n - 3 + k;
    if (t >= 0) a += xz[(size_t)t * (2 * DI) + d] * cw[d * 4 + k];
  }
  float s = a / (1.f + __expf(-a));
  xc[(size_t)n * DI + d] = s;
  xc_bf[(size_t)n * DI + d] = (__bf16)s;
}

// ---------------------------------------------------------------------------
// dt = softplus(dbc[:, :32] @ dt_proj_w^T + dt_proj_b). One row per block.
// ---------------------------------------------------------------------------
__global__ __launch_bounds__(256) void dt_k(const float* __restrict__ dbc,
                                            const float* __restrict__ wdt,
                                            const float* __restrict__ bdt,
                                            float* __restrict__ dt) {
  const int n = blockIdx.x, tid = threadIdx.x;
  __shared__ float r[DTR];
  if (tid < DTR) r[tid] = dbc[(size_t)n * NXP + tid];
  __syncthreads();
#pragma unroll
  for (int j = 0; j < DI / 256; ++j) {
    int d = tid + j * 256;
    float a = bdt[d];
#pragma unroll
    for (int k = 0; k < DTR; ++k) a += r[k] * wdt[d * DTR + k];
    dt[(size_t)n * DI + d] = (a > 20.f) ? a : log1pf(__expf(a));
  }
}

// ---------------------------------------------------------------------------
// Chunked parallel scan. Lane map: s = tid&15, d = blockIdx.x*16 + (tid>>4).
// ---------------------------------------------------------------------------
__global__ __launch_bounds__(256) void scan_p1_k(const float* __restrict__ dt,
                                                 const float* __restrict__ xc,
                                                 const float* __restrict__ dbc,
                                                 const float* __restrict__ A_log,
                                                 float* __restrict__ agg_a,
                                                 float* __restrict__ agg_h) {
  const int tid = threadIdx.x;
  const int s = tid & 15;
  const int d = blockIdx.x * 16 + (tid >> 4);
  const int chunk = blockIdx.y;
  const float Ads = -__expf(A_log[d * DSTATE + s]);
  float h = 0.f, aagg = 1.f;
  const int t0 = chunk * CLEN;
#pragma unroll 4
  for (int i = 0; i < CLEN; ++i) {
    const int t = t0 + i;
    float dtv = dt[(size_t)t * DI + d];
    float xv = xc[(size_t)t * DI + d];
    float Bv = dbc[(size_t)t * NXP + DTR + s];
    float a = __expf(dtv * Ads);
    h = a * h + (dtv * xv) * Bv;
    aagg *= a;
  }
  size_t o = (size_t)chunk * (DI * DSTATE) + blockIdx.x * 256 + tid;
  agg_a[o] = aagg;
  agg_h[o] = h;
}

__global__ __launch_bounds__(256) void scan_p2_k(const float* __restrict__ agg_a,
                                                 const float* __restrict__ agg_h,
                                                 float* __restrict__ init_h) {
  const int g = blockIdx.x * 256 + threadIdx.x;  // (d,s) flat, 16384 lanes
  float carry = 0.f;
  for (int c = 0; c < CHUNKS; ++c) {
    size_t o = (size_t)c * (DI * DSTATE) + g;
    init_h[o] = carry;
    carry = agg_a[o] * carry + agg_h[o];
  }
}

// Pass 3: replay with correct init, reduce y over 16 states, fuse D-skip + SiLU(z) gate.
__global__ __launch_bounds__(256) void scan_p3_k(const float* __restrict__ dt,
                                                 const float* __restrict__ xc,
                                                 const float* __restrict__ dbc,
                                                 const float* __restrict__ A_log,
                                                 const float* __restrict__ init_h,
                                                 const float* __restrict__ Dskip,
                                                 const float* __restrict__ xz,
                                                 __bf16* __restrict__ y_bf) {
  const int tid = threadIdx.x;
  const int s = tid & 15;
  const int d = blockIdx.x * 16 + (tid >> 4);
  const int chunk = blockIdx.y;
  const float Ads = -__expf(A_log[d * DSTATE + s]);
  float h = init_h[(size_t)chunk * (DI * DSTATE) + blockIdx.x * 256 + tid];
  const float Dd = Dskip[d];
  const int t0 = chunk * CLEN;
  for (int i = 0; i < CLEN; ++i) {
    const int t = t0 + i;
    float dtv = dt[(size_t)t * DI + d];
    float xv = xc[(size_t)t * DI + d];
    float Bv = dbc[(size_t)t * NXP + DTR + s];
    float Cv = dbc[(size_t)t * NXP + DTR + DSTATE + s];
    float a = __expf(dtv * Ads);
    h = a * h + (dtv * xv) * Bv;
    float y = h * Cv;
    y += __shfl_xor(y, 8);
    y += __shfl_xor(y, 4);
    y += __shfl_xor(y, 2);
    y += __shfl_xor(y, 1);
    if (s == 0) {
      float z = xz[(size_t)t * (2 * DI) + DI + d];
      float g = (y + xv * Dd) * (z / (1.f + __expf(-z)));
      y_bf[(size_t)t * DI + d] = (__bf16)g;
    }
  }
}

// ---------------------------------------------------------------------------
extern "C" void kernel_launch(void* const* d_in, const int* in_sizes, int n_in,
                              void* d_out, int out_size, void* d_ws, size_t ws_size,
                              hipStream_t stream) {
  (void)in_sizes; (void)n_in; (void)out_size; (void)ws_size;
  const float* feat     = (const float*)d_in[0];
  const int*   order    = (const int*)d_in[1];
  /* inverse d_in[2] unused: scatter via order (inverse∘order = id) */
  const float* ln1_w    = (const float*)d_in[3];
  const float* ln1_b    = (const float*)d_in[4];
  const float* in_proj  = (const float*)d_in[5];
  const float* conv_w   = (const float*)d_in[6];
  const float* conv_b   = (const float*)d_in[7];
  const float* x_proj   = (const float*)d_in[8];
  const float* dt_w     = (const float*)d_in[9];
  const float* dt_b     = (const float*)d_in[10];
  const float* A_log    = (const float*)d_in[11];
  const float* D_skip   = (const float*)d_in[12];
  const float* out_proj = (const float*)d_in[13];
  const float* ln2_w    = (const float*)d_in[14];
  const float* ln2_b    = (const float*)d_in[15];
  const float* ffn_w1   = (const float*)d_in[16];
  const float* ffn_b1   = (const float*)d_in[17];
  const float* ffn_w2   = (const float*)d_in[18];
  const float* ffn_b2   = (const float*)d_in[19];
  float* outp = (float*)d_out;

  char* ws = (char*)d_ws;
  size_t off = 0;
  auto alloc = [&](size_t bytes) -> void* {
    void* p = ws + off;
    off = (off + bytes + 255) & ~(size_t)255;
    return p;
  };
  float*  f      = (float*)alloc((size_t)NPTS * CDIM * 4);      // gathered residual
  float*  xz     = (float*)alloc((size_t)NPTS * 2 * DI * 4);    // in_proj out (xm|z)
  float*  xc     = (float*)alloc((size_t)NPTS * DI * 4);        // conv+silu
  float*  dbc    = (float*)alloc((size_t)NPTS * NXP * 4);       // x_proj out
  float*  dtb    = (float*)alloc((size_t)NPTS * DI * 4);        // softplus(dt)
  float*  x1     = (float*)alloc((size_t)NPTS * CDIM * 4);      // residual 2
  float*  agg_a  = (float*)alloc((size_t)CHUNKS * DI * DSTATE * 4);
  float*  agg_h  = (float*)alloc((size_t)CHUNKS * DI * DSTATE * 4);
  float*  init_h = (float*)alloc((size_t)CHUNKS * DI * DSTATE * 4);
  __bf16* xln_bf = (__bf16*)alloc((size_t)NPTS * CDIM * 2);     // reused as h2_bf
  __bf16* xc_bf  = (__bf16*)alloc((size_t)NPTS * DI * 2);       // reused as g_bf
  __bf16* y_bf   = (__bf16*)alloc((size_t)NPTS * DI * 2);
  __bf16* w_in   = (__bf16*)alloc((size_t)2 * DI * CDIM * 2);
  __bf16* w_xp   = (__bf16*)alloc((size_t)NXP * DI * 2);
  __bf16* w_out  = (__bf16*)alloc((size_t)CDIM * DI * 2);
  __bf16* w_f1t  = (__bf16*)alloc((size_t)MLPD * CDIM * 2);     // ffn_w1^T -> [MLP,C]
  __bf16* w_f2t  = (__bf16*)alloc((size_t)CDIM * MLPD * 2);     // ffn_w2^T -> [C,MLP]
  __bf16* h2_bf  = xln_bf;  // lifetime-disjoint reuse
  __bf16* g_bf   = xc_bf;   // lifetime-disjoint reuse

  // Weights -> bf16 (FFN weights transposed so every GEMM uses W[N,K]).
  cvt_bf_k<<<(2 * DI * CDIM + 255) / 256, 256, 0, stream>>>(in_proj, w_in, 2 * DI * CDIM);
  cvt_bf_k<<<(NXP * DI + 255) / 256, 256, 0, stream>>>(x_proj, w_xp, NXP * DI);
  cvt_bf_k<<<(CDIM * DI + 255) / 256, 256, 0, stream>>>(out_proj, w_out, CDIM * DI);
  cvt_bfT_k<<<(CDIM * MLPD + 255) / 256, 256, 0, stream>>>(ffn_w1, w_f1t, CDIM, MLPD);
  cvt_bfT_k<<<(MLPD * CDIM + 255) / 256, 256, 0, stream>>>(ffn_w2, w_f2t, MLPD, CDIM);

  // Gather + LN1 (stores residual f and bf16 LN output).
  ln512_k<true><<<NPTS, 256, 0, stream>>>(feat, order, ln1_w, ln1_b, f, xln_bf);

  // in_proj: (16384x512) x (2048x512)^T -> xz f32
  gemm_bf16_k<128, 128, 0><<<dim3(2 * DI / 128, NPTS / 128), 256, 0, stream>>>(
      xln_bf, w_in, CDIM, 2 * DI, xz, nullptr, nullptr, nullptr, nullptr);

  // depthwise causal conv + SiLU
  conv_silu_k<<<(NPTS * DI) / 256, 256, 0, stream>>>(xz, conv_w, conv_b, xc, xc_bf);

  // x_proj: (16384x1024) x (64x1024)^T -> dbc
  gemm_bf16_k<256, 64, 0><<<dim3(1, NPTS / 256), 256, 0, stream>>>(
      xc_bf, w_xp, DI, NXP, dbc, nullptr, nullptr, nullptr, nullptr);

  // dt = softplus(dbc[:,:32] @ dt_w^T + dt_b)
  dt_k<<<NPTS, 256, 0, stream>>>(dbc, dt_w, dt_b, dtb);

  // chunked parallel selective scan (fused gate into pass 3)
  scan_p1_k<<<dim3(DI / 16, CHUNKS), 256, 0, stream>>>(dtb, xc, dbc, A_log, agg_a, agg_h);
  scan_p2_k<<<(DI * DSTATE) / 256, 256, 0, stream>>>(agg_a, agg_h, init_h);
  scan_p3_k<<<dim3(DI / 16, CHUNKS), 256, 0, stream>>>(dtb, xc, dbc, A_log, init_h,
                                                       D_skip, xz, y_bf);

  // out_proj + residual: x1 = f + y @ out_proj^T
  gemm_bf16_k<128, 128, 1><<<dim3(CDIM / 128, NPTS / 128), 256, 0, stream>>>(
      y_bf, w_out, DI, CDIM, x1, nullptr, f, nullptr, nullptr);

  // LN2
  ln512_k<false><<<NPTS, 256, 0, stream>>>(x1, nullptr, ln2_w, ln2_b, nullptr, h2_bf);

  // FFN1 + exact GELU -> bf16
  gemm_bf16_k<128, 128, 2><<<dim3(MLPD / 128, NPTS / 128), 256, 0, stream>>>(
      h2_bf, w_f1t, CDIM, MLPD, nullptr, g_bf, nullptr, ffn_b1, nullptr);

  // FFN2 + bias + residual + fused scatter: d_out[order[n]] = x1[n] + ...
  gemm_bf16_k<128, 128, 3><<<dim3(CDIM / 128, NPTS / 128), 256, 0, stream>>>(
      g_bf, w_f2t, MLPD, CDIM, outp, nullptr, x1, ffn_b2, order);
}
